// QuantumSelfAttention_24068996727396
// MI455X (gfx1250) — matrix operand order repaired and checked by reading, not compiled, not run on hardware
//
#include <hip/hip_runtime.h>

typedef __attribute__((ext_vector_type(16))) _Float16 v16h;
typedef __attribute__((ext_vector_type(8)))  _Float16 v8h;
typedef __attribute__((ext_vector_type(8)))  float    v8f;

#define BB 16
#define SS 512
#define EE 1024
#define QQ 8
#define LL 4

// ---------------- elementwise f32 -> f16 ----------------
__global__ void k_cvt_f16(const float* __restrict__ in, _Float16* __restrict__ out, int n) {
  int i = blockIdx.x * blockDim.x + threadIdx.x;
  if (i < n) out[i] = (_Float16)in[i];
}

// ---------------- transpose-convert W[K][N] f32 -> Wt[N][K] f16 ----------------
__global__ void k_cvtT_f16(const float* __restrict__ in, _Float16* __restrict__ out,
                           int rows, int cols) {
  int i = blockIdx.x * blockDim.x + threadIdx.x;
  if (i >= rows * cols) return;
  int r = i / cols, c = i % cols;
  out[(size_t)c * rows + r] = (_Float16)in[i];
}

// ---------------- qubit states: x = X@Wq + bq, then composed SU(2) layers ----------------
// U[bs][q][4] = (Re a, Im a, Re b, Im b)
__global__ void k_states(const float* __restrict__ X, const float* __restrict__ Wq,
                         const float* __restrict__ bq, const float* __restrict__ theta,
                         float* __restrict__ U) {
  int t  = blockIdx.x * blockDim.x + threadIdx.x;   // B*S*Q threads exactly
  int q  = t & (QQ - 1);
  int bs = t >> 3;
  const float* xr = X + (size_t)bs * EE;
  float acc = 0.f;
  #pragma unroll 4
  for (int e = 0; e < EE; ++e) acc += xr[e] * Wq[e * QQ + q];
  float x = acc + bq[q];
  float ar = cosf(0.5f * x), ai = 0.f;
  float br = sinf(0.5f * x), bi = 0.f;
  #pragma unroll
  for (int l = 0; l < LL; ++l) {
    float t0 = 0.5f * theta[(l * QQ + q) * 2 + 0];
    float t1 = 0.5f * theta[(l * QQ + q) * 2 + 1];
    float c = cosf(t0), s = sinf(t0);
    float nar = c * ar - s * br, nai = c * ai - s * bi;
    float nbr = s * ar + c * br, nbi = s * ai + c * bi;
    float cp = cosf(t1), sp = sinf(t1);
    ar = nar * cp + nai * sp;  ai = nai * cp - nar * sp;   // a *= exp(-i t1)
    br = nbr * cp - nbi * sp;  bi = nbi * cp + nbr * sp;   // b *= exp(+i t1)
  }
  float* u = U + (size_t)t * 4;
  u[0] = ar; u[1] = ai; u[2] = br; u[3] = bi;
}

// ---------------- generic batched WMMA GEMM: C[M,N] = A[M,K] * Bt[N,K]^T (+bias) ----------------
// emode 0: f32 row-major (ldc=N stride), 1: f16 row-major, 2: f16 transposed Ct[n*ldc + m]
__global__ void __launch_bounds__(256)
k_gemm(const _Float16* __restrict__ A, size_t sA, int lda,
       const _Float16* __restrict__ Bt, size_t sB, int ldb,
       const float* __restrict__ bias,
       float* __restrict__ C32, _Float16* __restrict__ C16, size_t sC,
       int K, int emode, int ldc) {
  int lane = threadIdx.x & 31;
  int wave = threadIdx.x >> 5;
  int hf   = lane >> 4;
  int rw   = lane & 15;
  int b    = blockIdx.z;
  int tm   = blockIdx.x * 32;
  int tn   = (blockIdx.y * 8 + wave) * 32;

  const _Float16* Ab = A  + (size_t)b * sA;
  const _Float16* Bb = Bt + (size_t)b * sB;

  const _Float16* a0p = Ab + (size_t)(tm + rw)      * lda;
  const _Float16* a1p = Ab + (size_t)(tm + 16 + rw) * lda;
  const _Float16* b0p = Bb + (size_t)(tn + rw)      * ldb;
  const _Float16* b1p = Bb + (size_t)(tn + 16 + rw) * ldb;
  const int kA1 = 8 * hf, kA2 = 16 + 8 * hf, kB = 16 * hf;

  v8f c00 = {}, c01 = {}, c10 = {}, c11 = {};

  for (int k = 0; k < K; k += 32) {
    v8h al0 = *(const v8h*)(a0p + k + kA1);
    v8h ah0 = *(const v8h*)(a0p + k + kA2);
    v8h al1 = *(const v8h*)(a1p + k + kA1);
    v8h ah1 = *(const v8h*)(a1p + k + kA2);
    v8h bl0 = *(const v8h*)(b0p + k + kA1);
    v8h bh0 = *(const v8h*)(b0p + k + kA2);
    v8h bl1 = *(const v8h*)(b1p + k + kA1);
    v8h bh1 = *(const v8h*)(b1p + k + kA2);
    v16h a0 = __builtin_shufflevector(al0, ah0, 0,1,2,3,4,5,6,7,8,9,10,11,12,13,14,15);
    v16h a1 = __builtin_shufflevector(al1, ah1, 0,1,2,3,4,5,6,7,8,9,10,11,12,13,14,15);
    v16h b0 = __builtin_shufflevector(bl0, bh0, 0,1,2,3,4,5,6,7,8,9,10,11,12,13,14,15);
    v16h b1 = __builtin_shufflevector(bl1, bh1, 0,1,2,3,4,5,6,7,8,9,10,11,12,13,14,15);
    c00 = __builtin_amdgcn_wmma_f32_16x16x32_f16(false, a0, false, b0, (short)0, c00, false, false);
    c01 = __builtin_amdgcn_wmma_f32_16x16x32_f16(false, a0, false, b1, (short)0, c01, false, false);
    c10 = __builtin_amdgcn_wmma_f32_16x16x32_f16(false, a1, false, b0, (short)0, c10, false, false);
    c11 = __builtin_amdgcn_wmma_f32_16x16x32_f16(false, a1, false, b1, (short)0, c11, false, false);
  }

  int ms0 = tm + 8 * hf, ms1 = tm + 16 + 8 * hf;
  int nc0 = tn + rw,     nc1 = tn + 16 + rw;
  float bi0 = bias ? bias[nc0] : 0.f;
  float bi1 = bias ? bias[nc1] : 0.f;

  if (emode == 0) {
    float* Cb = C32 + (size_t)b * sC;
    #pragma unroll
    for (int r = 0; r < 8; ++r) {
      Cb[(size_t)(ms0 + r) * ldc + nc0] = c00[r] + bi0;
      Cb[(size_t)(ms0 + r) * ldc + nc1] = c01[r] + bi1;
      Cb[(size_t)(ms1 + r) * ldc + nc0] = c10[r] + bi0;
      Cb[(size_t)(ms1 + r) * ldc + nc1] = c11[r] + bi1;
    }
  } else if (emode == 1) {
    _Float16* Cb = C16 + (size_t)b * sC;
    #pragma unroll
    for (int r = 0; r < 8; ++r) {
      Cb[(size_t)(ms0 + r) * ldc + nc0] = (_Float16)c00[r];
      Cb[(size_t)(ms0 + r) * ldc + nc1] = (_Float16)c01[r];
      Cb[(size_t)(ms1 + r) * ldc + nc0] = (_Float16)c10[r];
      Cb[(size_t)(ms1 + r) * ldc + nc1] = (_Float16)c11[r];
    }
  } else {
    // transposed f16: Ct[n*ldc + m], 8 consecutive m per lane -> one packed 16B store each
    _Float16* Cb = C16 + (size_t)b * sC;
    v8h p00, p01, p10, p11;
    #pragma unroll
    for (int r = 0; r < 8; ++r) {
      p00[r] = (_Float16)(c00[r] + bi0);
      p01[r] = (_Float16)(c01[r] + bi1);
      p10[r] = (_Float16)(c10[r] + bi0);
      p11[r] = (_Float16)(c11[r] + bi1);
    }
    *(v8h*)(Cb + (size_t)nc0 * ldc + ms0) = p00;
    *(v8h*)(Cb + (size_t)nc1 * ldc + ms0) = p01;
    *(v8h*)(Cb + (size_t)nc0 * ldc + ms1) = p10;
    *(v8h*)(Cb + (size_t)nc1 * ldc + ms1) = p11;
  }
}

// ---------------- fidelity kernel K + interference + softmax ----------------
// one block (256 threads) per (b, i) row; each thread covers j = t, t+256
__global__ void __launch_bounds__(256)
k_attn(const float* __restrict__ U, const float* __restrict__ phi,
       float* __restrict__ attnF32, _Float16* __restrict__ attnF16) {
  __shared__ float ui[32];
  __shared__ float red[256];
  int b = blockIdx.x >> 9;
  int i = blockIdx.x & (SS - 1);
  int t = threadIdx.x;
  const float* Ub = U + (size_t)b * SS * 32;
  if (t < 32) ui[t] = Ub[(size_t)i * 32 + t];
  __syncthreads();
  float ph_i = phi[i];

  float kv[2];
  #pragma unroll
  for (int p = 0; p < 2; ++p) {
    int j = t + p * 256;
    const float* uj = Ub + (size_t)j * 32;
    float prod = 1.f;
    #pragma unroll
    for (int q = 0; q < QQ; ++q) {
      float u0 = ui[q*4], u1 = ui[q*4+1], u2 = ui[q*4+2], u3 = ui[q*4+3];
      float w0 = uj[q*4], w1 = uj[q*4+1], w2 = uj[q*4+2], w3 = uj[q*4+3];
      float re = u0*w0 + u1*w1 + u2*w2 + u3*w3;
      float im = u0*w1 - u1*w0 + u2*w3 - u3*w2;
      prod *= re*re + im*im;
    }
    kv[p] = prod + cosf(ph_i - phi[j]);
  }

  float m = fmaxf(kv[0], kv[1]);
  red[t] = m; __syncthreads();
  for (int s2 = 128; s2 > 0; s2 >>= 1) {
    if (t < s2) red[t] = fmaxf(red[t], red[t + s2]);
    __syncthreads();
  }
  float mx = red[0]; __syncthreads();
  float e0 = __expf(kv[0] - mx), e1 = __expf(kv[1] - mx);
  red[t] = e0 + e1; __syncthreads();
  for (int s2 = 128; s2 > 0; s2 >>= 1) {
    if (t < s2) red[t] += red[t + s2];
    __syncthreads();
  }
  float inv = 1.f / red[0];
  size_t base = ((size_t)b * SS + i) * SS;
  float a0 = e0 * inv, a1 = e1 * inv;
  attnF32[base + t]       = a0;  attnF16[base + t]       = (_Float16)a0;
  attnF32[base + t + 256] = a1;  attnF16[base + t + 256] = (_Float16)a1;
}

extern "C" void kernel_launch(void* const* d_in, const int* in_sizes, int n_in,
                              void* d_out, int out_size, void* d_ws, size_t ws_size,
                              hipStream_t stream) {
  const float* X     = (const float*)d_in[0];
  const float* Wq    = (const float*)d_in[1];
  const float* bq    = (const float*)d_in[2];
  const float* Wv    = (const float*)d_in[3];
  const float* bv    = (const float*)d_in[4];
  const float* Wo    = (const float*)d_in[5];
  const float* bo    = (const float*)d_in[6];
  const float* theta = (const float*)d_in[7];
  const float* phi   = (const float*)d_in[8];

  float* outF32  = (float*)d_out;                       // [B,S,E]
  float* attnF32 = (float*)d_out + (size_t)BB * SS * EE; // [B,S,S]

  // workspace layout (bytes, all 16B-aligned)
  char* ws = (char*)d_ws;
  _Float16* Xf16   = (_Float16*)(ws);                       // 16 MB
  _Float16* WvT    = (_Float16*)(ws + 16777216);            //  2 MB
  _Float16* WoT    = (_Float16*)(ws + 18874368);            //  2 MB
  _Float16* Vt     = (_Float16*)(ws + 20971520);            // 16 MB [b][e][s]
  _Float16* attn16 = (_Float16*)(ws + 37748736);            //  8 MB
  _Float16* T16    = (_Float16*)(ws + 46137344);            // 16 MB [b][s][e]
  float*    U      = (float*)   (ws + 62914560);            //  1 MB [b][s][q][4]

  const int nX = BB * SS * EE;        // 8388608
  const int nW = EE * EE;             // 1048576
  const size_t sME = (size_t)SS * EE; // 524288
  const size_t sMS = (size_t)SS * SS; // 262144

  // 1) conversions
  k_cvt_f16 <<<nX / 256, 256, 0, stream>>>(X, Xf16, nX);
  k_cvtT_f16<<<nW / 256, 256, 0, stream>>>(Wv, WvT, EE, EE);
  k_cvtT_f16<<<nW / 256, 256, 0, stream>>>(Wo, WoT, EE, EE);

  // 2) qubit states
  k_states<<<(BB * SS * QQ) / 256, 256, 0, stream>>>(X, Wq, bq, theta, U);

  // 3) V = X@Wv + bv, stored transposed f16: Vt[b][e][s]
  k_gemm<<<dim3(SS / 32, EE / 256, BB), 256, 0, stream>>>(
      Xf16, sME, EE, WvT, 0, EE, bv, nullptr, Vt, sME, EE, /*emode=*/2, /*ldc=*/SS);

  // 4) fidelity + softmax -> attn (f32 output + f16 operand)
  k_attn<<<BB * SS, 256, 0, stream>>>(U, phi, attnF32, attn16);

  // 5) T = attn @ V   (Bt = Vt)
  k_gemm<<<dim3(SS / 32, EE / 256, BB), 256, 0, stream>>>(
      attn16, sMS, SS, Vt, sME, SS, nullptr, nullptr, T16, sME, SS, /*emode=*/1, /*ldc=*/EE);

  // 6) out = T @ Wo + bo  (f32)
  k_gemm<<<dim3(SS / 32, EE / 256, BB), 256, 0, stream>>>(
      T16, sME, EE, WoT, 0, EE, bo, outF32, nullptr, sME, EE, /*emode=*/0, /*ldc=*/EE);
}
